// RetrievalCrossAttention_22204980920631
// MI455X (gfx1250) — matrix operand order repaired and checked by reading, not compiled, not run on hardware
//
#include <hip/hip_runtime.h>
#include <hip/hip_bf16.h>
#include <math.h>
#include <stdint.h>

// ---------------------------------------------------------------------------
// Types
// ---------------------------------------------------------------------------
typedef __attribute__((ext_vector_type(16))) _Float16 v16h;
typedef __attribute__((ext_vector_type(8)))  _Float16 v8h;
typedef __attribute__((ext_vector_type(4)))  _Float16 v4h;
typedef __attribute__((ext_vector_type(8)))  float    v8f;

static __device__ __forceinline__ int imin(int a, int b) { return a < b ? a : b; }

// ---------------------------------------------------------------------------
// CDNA5 async global->LDS staging via inline asm (ASYNCcnt-tracked).
// VDST holds the LDS byte offset = low 32 bits of the flat pointer to a
// __shared__ object (ISA 10.2).
// ---------------------------------------------------------------------------
#define USE_ASYNC_LDS 1

static __device__ __forceinline__ void async_copy_b128(const void* gptr, void* lptr) {
    asm volatile("global_load_async_to_lds_b128 %0, %1, off"
                 :
                 : "v"((uint32_t)(uintptr_t)lptr),
                   "v"((unsigned long long)(uintptr_t)gptr)
                 : "memory");
}
// Wait until at most the newest stage-group (8 async issues/wave) remains.
static __device__ __forceinline__ void async_wait_le8() {
    asm volatile("s_wait_asynccnt 0x8" ::: "memory");
}
static __device__ __forceinline__ void async_wait0() {
    asm volatile("s_wait_asynccnt 0x0" ::: "memory");
}

// ---------------------------------------------------------------------------
// f32 -> f16 conversion (vectorized, n must be a multiple of 4)
// ---------------------------------------------------------------------------
__global__ __launch_bounds__(256)
void f32_to_f16_vec4(const float* __restrict__ src, _Float16* __restrict__ dst, int n4) {
    int i = blockIdx.x * blockDim.x + threadIdx.x;
    if (i < n4) {
        float4 v = ((const float4*)src)[i];
        v4h o;
        o[0] = (_Float16)v.x; o[1] = (_Float16)v.y;
        o[2] = (_Float16)v.z; o[3] = (_Float16)v.w;
        ((v4h*)dst)[i] = o;
    }
}

// ---------------------------------------------------------------------------
// LayerNorm over D=1024, one row per 256-thread block, f32 in -> f16 out
// ---------------------------------------------------------------------------
__global__ __launch_bounds__(256)
void layernorm_f32_to_f16(const float* __restrict__ x, const float* __restrict__ w,
                          const float* __restrict__ b, _Float16* __restrict__ y) {
    const int row = blockIdx.x;
    const int tid = threadIdx.x;
    const float* xr = x + (size_t)row * 1024 + tid * 4;
    float4 v = *(const float4*)xr;

    __shared__ float rs[256], rq[256];
    rs[tid] = v.x + v.y + v.z + v.w;
    rq[tid] = v.x*v.x + v.y*v.y + v.z*v.z + v.w*v.w;
    __syncthreads();
    for (int off = 128; off > 0; off >>= 1) {
        if (tid < off) { rs[tid] += rs[tid + off]; rq[tid] += rq[tid + off]; }
        __syncthreads();
    }
    const float mean = rs[0] * (1.0f / 1024.0f);
    const float var  = rq[0] * (1.0f / 1024.0f) - mean * mean;
    const float inv  = rsqrtf(var + 1e-5f);

    float4 wg = *(const float4*)(w + tid * 4);
    float4 bg = *(const float4*)(b + tid * 4);
    v4h o;
    o[0] = (_Float16)((v.x - mean) * inv * wg.x + bg.x);
    o[1] = (_Float16)((v.y - mean) * inv * wg.y + bg.y);
    o[2] = (_Float16)((v.z - mean) * inv * wg.z + bg.z);
    o[3] = (_Float16)((v.w - mean) * inv * wg.w + bg.w);
    *(v4h*)(y + (size_t)row * 1024 + tid * 4) = o;
}

// ---------------------------------------------------------------------------
// WMMA GEMM:  C[M,N] = A[M,K](f16, row-major) x W[N,K](f16, row-major)^T
//             + bias[N] (+ epilogue)
// Block: 256 threads = 8 wave32.  Tile 128(M) x 128(N) x 64(K), LDS triple-
// buffered with async global->LDS staging (newest stage stays in flight
// across the barrier).  Waves arranged 2(M)x4(N); each wave owns 64x32 and
// runs 2 K-sub-steps => 16 v_wmma per wave per barrier.
// ---------------------------------------------------------------------------
enum { EPI_STORE = 0, EPI_GELU = 1, EPI_RES = 2 };

#define BM 128
#define BN 128
#define BK 64
#define LPAD 8   // f16 elements of padding (16B) per LDS row

template <typename OutT, int EPI>
__global__ __launch_bounds__(256)
void gemm_f16_wmma(const _Float16* __restrict__ A, const _Float16* __restrict__ W,
                   const float* __restrict__ bias, const float* __restrict__ res,
                   OutT* __restrict__ C, int M, int N, int K) {
    __shared__ __attribute__((aligned(16))) _Float16 As[3][BM][BK + LPAD];
    __shared__ __attribute__((aligned(16))) _Float16 Bs[3][BN][BK + LPAD];

    const int tid  = threadIdx.x;
    const int lane = tid & 31;
    const int wave = tid >> 5;
    const int ws   = wave >> 2;  // 64-row strip within block (0..1)
    const int wn   = wave & 3;   // 32-col strip within block (0..3)
    const int m0   = blockIdx.x * BM;
    const int n0   = blockIdx.y * BN;

    // Staging: 256 threads; thread -> (row = tid/2, 32-f16 half = tid&1) for
    // both the 128x64 A tile and the 128x64 B tile (4 x b128 each).
    const int sr = tid >> 1;
    const int sk = (tid & 1) * 32;
    const size_t aRow = (size_t)imin(m0 + sr, M - 1);
    const size_t bRow = (size_t)imin(n0 + sr, N - 1);
    const _Float16* aBase = A + aRow * K + sk;
    const _Float16* bBase = W + bRow * K + sk;

    // Fragment lane mapping (ISA 7.12.2)
    const int lm  = lane & 15;
    const int klo = (lane < 16) ? 0 : 8;    // A frag: {klo..+7, klo+16..+23}
    const int khi = (lane < 16) ? 0 : 8;    // B frag: {khi..+7, khi+16..+23}

    v8f acc[4][2] = {};

    auto stage = [&](int sbuf, int kt) {  // 8 async issues per thread
        const _Float16* ag = aBase + kt * BK;
        const _Float16* bg = bBase + kt * BK;
#if USE_ASYNC_LDS
#pragma unroll
        for (int c = 0; c < 4; ++c) {
            async_copy_b128(ag + c * 8, &As[sbuf][sr][sk + c * 8]);
            async_copy_b128(bg + c * 8, &Bs[sbuf][sr][sk + c * 8]);
        }
#else
#pragma unroll
        for (int c = 0; c < 4; ++c) {
            *(v8h*)&As[sbuf][sr][sk + c * 8] = *(const v8h*)(ag + c * 8);
            *(v8h*)&Bs[sbuf][sr][sk + c * 8] = *(const v8h*)(bg + c * 8);
        }
#endif
    };

    const int nk = K / BK;
    stage(0, 0);
    if (nk > 1) stage(1, 1);

    for (int kt = 0; kt < nk; ++kt) {
        const int buf = kt % 3;
#if USE_ASYNC_LDS
        // Oldest outstanding stage (tile kt) must be complete; the newest
        // (tile kt+1, 8 issues) may stay in flight across the barrier.
        if (kt + 1 < nk) async_wait_le8();
        else             async_wait0();
#endif
        __syncthreads();
        if (kt + 2 < nk) stage((kt + 2) % 3, kt + 2);

#pragma unroll
        for (int ks = 0; ks < 2; ++ks) {  // two 32-wide K sub-steps
            const int kb = ks * 32;
            v16h af[4], bf[2];
#pragma unroll
            for (int mi = 0; mi < 4; ++mi) {
                const _Float16* p = &As[buf][ws * 64 + mi * 16 + lm][kb + klo];
                v8h lo = *(const v8h*)p;
                v8h hi = *(const v8h*)(p + 16);
                af[mi] = __builtin_shufflevector(lo, hi, 0,1,2,3,4,5,6,7,8,9,10,11,12,13,14,15);
            }
#pragma unroll
            for (int ni = 0; ni < 2; ++ni) {
                const _Float16* p = &Bs[buf][wn * 32 + ni * 16 + lm][kb + khi];
                v8h lo = *(const v8h*)p;
                v8h hi = *(const v8h*)(p + 16);
                bf[ni] = __builtin_shufflevector(lo, hi, 0,1,2,3,4,5,6,7,8,9,10,11,12,13,14,15);
            }
#pragma unroll
            for (int mi = 0; mi < 4; ++mi)
#pragma unroll
                for (int ni = 0; ni < 2; ++ni)
                    acc[mi][ni] = __builtin_amdgcn_wmma_f32_16x16x32_f16(
                        false, af[mi], false, bf[ni], (short)0, acc[mi][ni], false, false);
        }
    }

    // ---- epilogue: C VGPR j -> (row j + 8*(lane>=16), col lane%16) ----
    const int rbase = (lane >> 4) * 8;
#pragma unroll
    for (int mi = 0; mi < 4; ++mi) {
#pragma unroll
        for (int ni = 0; ni < 2; ++ni) {
            const int gc = n0 + wn * 32 + ni * 16 + lm;
            const float bn = bias ? bias[gc] : 0.0f;
#pragma unroll
            for (int j = 0; j < 8; ++j) {
                const int gr = m0 + ws * 64 + mi * 16 + rbase + j;
                if (gr < M) {
                    float v = acc[mi][ni][j] + bn;
                    if (EPI == EPI_GELU)
                        v = 0.5f * v * (1.0f + erff(v * 0.70710678118654752f));
                    if (EPI == EPI_RES)
                        v += res[(size_t)gr * N + gc];
                    C[(size_t)gr * N + gc] = (OutT)v;
                }
            }
        }
    }
}

// ---------------------------------------------------------------------------
// Single-query cross attention. One block per (b,h).
// kvp: [B*2048, 2048] f16; cols 0..1023 = K-proj, 1024..2047 = V-proj.
// qp : [B, 1024] f16.  ctx out: [B, 1024] f16.
// ---------------------------------------------------------------------------
__global__ __launch_bounds__(256)
void cross_attn_kernel(const _Float16* __restrict__ qp,
                       const _Float16* __restrict__ kvp,
                       _Float16* __restrict__ ctx) {
    const int b   = blockIdx.x >> 4;
    const int h   = blockIdx.x & 15;
    const int tid = threadIdx.x;

    __shared__ float sc[2048];
    __shared__ float red[256];
    __shared__ float qv[64];

    if (tid < 64) qv[tid] = (float)qp[b * 1024 + h * 64 + tid];
    __syncthreads();

    const _Float16* kbase = kvp + ((size_t)b * 2048) * 2048 + h * 64;

    // scores = q . k / sqrt(64)   (8 x b128 loads per key)
    float lmax = -1e30f;
#pragma unroll
    for (int it = 0; it < 8; ++it) {
        const int kk = tid + it * 256;
        const v8h* kr = (const v8h*)(kbase + (size_t)kk * 2048);
        float s = 0.0f;
#pragma unroll
        for (int c = 0; c < 8; ++c) {
            v8h kd = kr[c];
#pragma unroll
            for (int j = 0; j < 8; ++j) s += qv[c * 8 + j] * (float)kd[j];
        }
        s *= 0.125f;
        sc[kk] = s;
        lmax = fmaxf(lmax, s);
    }
    red[tid] = lmax;
    __syncthreads();
    for (int off = 128; off > 0; off >>= 1) {
        if (tid < off) red[tid] = fmaxf(red[tid], red[tid + off]);
        __syncthreads();
    }
    const float mx = red[0];
    __syncthreads();

    float lsum = 0.0f;
#pragma unroll
    for (int it = 0; it < 8; ++it) {
        const int kk = tid + it * 256;
        const float p = __expf(sc[kk] - mx);
        sc[kk] = p;
        lsum += p;
    }
    red[tid] = lsum;
    __syncthreads();
    for (int off = 128; off > 0; off >>= 1) {
        if (tid < off) red[tid] += red[tid + off];
        __syncthreads();
    }
    const float denom = red[0];
    __syncthreads();

    // ctx[d] = sum_k p[k] * v[k][d] ; thread = (kgroup, d)
    const int d   = tid & 63;
    const int grp = tid >> 6;  // 4 groups x 512 keys
    const _Float16* vbase = kvp + ((size_t)b * 2048) * 2048 + 1024 + h * 64 + d;
    float a = 0.0f;
    for (int kk = grp * 512; kk < grp * 512 + 512; ++kk)
        a += sc[kk] * (float)vbase[(size_t)kk * 2048];
    red[tid] = a;
    __syncthreads();
    if (tid < 64) {
        const float t = red[tid] + red[tid + 64] + red[tid + 128] + red[tid + 192];
        ctx[b * 1024 + h * 64 + tid] = (_Float16)(t / denom);
    }
}

// ---------------------------------------------------------------------------
// Launcher
// ---------------------------------------------------------------------------
extern "C" void kernel_launch(void* const* d_in, const int* in_sizes, int n_in,
                              void* d_out, int out_size, void* d_ws, size_t ws_size,
                              hipStream_t stream) {
    (void)in_sizes; (void)n_in; (void)out_size; (void)ws_size;

    const float* q     = (const float*)d_in[0];
    const float* kv    = (const float*)d_in[1];
    const float* nq_w  = (const float*)d_in[2];
    const float* nq_b  = (const float*)d_in[3];
    const float* nkv_w = (const float*)d_in[4];
    const float* nkv_b = (const float*)d_in[5];
    const float* in_w  = (const float*)d_in[6];
    const float* in_b  = (const float*)d_in[7];
    const float* out_w = (const float*)d_in[8];
    const float* out_b = (const float*)d_in[9];
    const float* fln_w = (const float*)d_in[10];
    const float* fln_b = (const float*)d_in[11];
    const float* w1    = (const float*)d_in[12];
    const float* b1    = (const float*)d_in[13];
    const float* w2    = (const float*)d_in[14];
    const float* b2    = (const float*)d_in[15];
    float* out = (float*)d_out;

    // Workspace carve-up (~430 MB)
    size_t o = 0;
    auto take = [&](size_t bytes) -> char* {
        char* p = (char*)d_ws + o;
        o += (bytes + 255) & ~(size_t)255;
        return p;
    };
    _Float16* kvn_h = (_Float16*)take(65536ull * 1024 * 2);  // LN(kv) f16
    _Float16* kvp_h = (_Float16*)take(65536ull * 2048 * 2);  // K|V projections
    _Float16* wih   = (_Float16*)take(3072ull * 1024 * 2);   // in_w f16
    _Float16* woh   = (_Float16*)take(1024ull * 1024 * 2);   // out_w f16
    _Float16* w1h   = (_Float16*)take(4096ull * 1024 * 2);   // w1 f16
    _Float16* w2h   = (_Float16*)take(1024ull * 4096 * 2);   // w2 f16
    _Float16* qn_h  = (_Float16*)take(32 * 1024 * 2);
    _Float16* qp_h  = (_Float16*)take(32 * 1024 * 2);
    _Float16* ctx_h = (_Float16*)take(32 * 1024 * 2);
    float*    x_f   = (float*)   take(32 * 1024 * 4);
    _Float16* hn_h  = (_Float16*)take(32 * 1024 * 2);
    _Float16* h1_h  = (_Float16*)take(32 * 4096 * 2);

    // 1) weights -> f16
    f32_to_f16_vec4<<<(3072 * 1024 / 4 + 255) / 256, 256, 0, stream>>>(in_w,  wih, 3072 * 1024 / 4);
    f32_to_f16_vec4<<<(1024 * 1024 / 4 + 255) / 256, 256, 0, stream>>>(out_w, woh, 1024 * 1024 / 4);
    f32_to_f16_vec4<<<(4096 * 1024 / 4 + 255) / 256, 256, 0, stream>>>(w1,    w1h, 4096 * 1024 / 4);
    f32_to_f16_vec4<<<(1024 * 4096 / 4 + 255) / 256, 256, 0, stream>>>(w2,    w2h, 1024 * 4096 / 4);

    // 2) layernorms
    layernorm_f32_to_f16<<<65536, 256, 0, stream>>>(kv, nkv_w, nkv_b, kvn_h);
    layernorm_f32_to_f16<<<32,    256, 0, stream>>>(q,  nq_w,  nq_b,  qn_h);

    // 3) projections (the K/V one is the hot GEMM: 275 GFLOP)
    gemm_f16_wmma<_Float16, EPI_STORE><<<dim3(1, 1024 / BN), 256, 0, stream>>>(
        qn_h, wih, in_b, nullptr, qp_h, 32, 1024, 1024);
    gemm_f16_wmma<_Float16, EPI_STORE><<<dim3(65536 / BM, 2048 / BN), 256, 0, stream>>>(
        kvn_h, wih + 1024 * 1024, in_b + 1024, nullptr, kvp_h, 65536, 2048, 1024);

    // 4) attention (one block per (b,h))
    cross_attn_kernel<<<32 * 16, 256, 0, stream>>>(qp_h, kvp_h, ctx_h);

    // 5) output projection + residual with q
    gemm_f16_wmma<float, EPI_RES><<<dim3(1, 1024 / BN), 256, 0, stream>>>(
        ctx_h, woh, out_b, q, x_f, 32, 1024, 1024);

    // 6) FFN
    layernorm_f32_to_f16<<<32, 256, 0, stream>>>(x_f, fln_w, fln_b, hn_h);
    gemm_f16_wmma<_Float16, EPI_GELU><<<dim3(1, 4096 / BN), 256, 0, stream>>>(
        hn_h, w1h, b1, nullptr, h1_h, 32, 4096, 1024);
    gemm_f16_wmma<float, EPI_RES><<<dim3(1, 1024 / BN), 256, 0, stream>>>(
        h1_h, w2h, b2, x_f, out, 32, 1024, 4096);
}
